// MultiHeadAttention_3015067042376
// MI455X (gfx1250) — matrix operand order repaired and checked by reading, not compiled, not run on hardware
//
#include <hip/hip_runtime.h>
#include <hip/hip_bf16.h>

// ---------------------------------------------------------------------------
// Causal multi-head attention for MI455X (gfx1250), bf16 WMMA path.
// B=4, S=2048, D=1024, H=16, HD=64.
//  - x and weights converted once to bf16 (weights transposed so WMMA A and B
//    fragments are two contiguous 16B loads per lane).
//  - Q/K/V projections + output projection: WMMA GEMM, 32x64 tile per wave,
//    K-loop unrolled 4x for deep load/WMMA software pipelining.
//  - 1/sqrt(HD) folded into the Q projection (no per-score multiply).
//  - Flash attention: one wave per (b,h,16-query tile), online softmax with
//    16-lane shfl_xor reductions, P relayout through LDS, V pre-transposed.
// ---------------------------------------------------------------------------

typedef __bf16 bf16_t;
typedef __attribute__((ext_vector_type(16))) __bf16 v16bf;
typedef __attribute__((ext_vector_type(8)))  float  v8f;

constexpr int BB  = 4;
constexpr int SS  = 2048;
constexpr int DD  = 1024;
constexpr int HH  = 16;
constexpr int HDI = 64;
constexpr float SCALE = 0.125f; // 1/sqrt(HD)

// float -> bf16 with round-to-nearest-even
__device__ __forceinline__ bf16_t f2bf(float f) {
  union { float f; unsigned u; } v; v.f = f;
  unsigned r = v.u + 0x7FFFu + ((v.u >> 16) & 1u);
  union { unsigned short s; bf16_t b; } o; o.s = (unsigned short)(r >> 16);
  return o.b;
}

// ---- WMMA fragment loaders (wave32, 16x16x32 bf16 shapes) ------------------
// A fragment: 16x32 (MxK), row-major source with leading dim `ld`.
// ISA layout: lanes 0-15 hold M=lane, K={0..7,16..23}; lanes 16-31 hold
// M=lane-16, K={8..15,24..31}.  -> two 16B contiguous loads per lane.
__device__ __forceinline__ v16bf load_a_frag(const bf16_t* base, int ld,
                                             int m0, int k0) {
  int lane = threadIdx.x & 31;
  const bf16_t* row = base + (size_t)(m0 + (lane & 15)) * ld + k0
                    + ((lane < 16) ? 0 : 8);
  union { uint4 u[2]; v16bf v; } t;
  t.u[0] = *(const uint4*)(row);
  t.u[1] = *(const uint4*)(row + 16);
  return t.v;
}

// B fragment: 32x16 (KxN) loaded from an N-major ("transposed") matrix
// BT[n][k] with leading dim `ld`.  Lanes 0-15: N=lane, K={0..7,16..23};
// lanes 16-31: N=lane-16, K={8..15,24..31}.  -> two 16B contiguous loads per lane.
__device__ __forceinline__ v16bf load_b_frag_t(const bf16_t* baseT, int ld,
                                               int k0, int n0) {
  int lane = threadIdx.x & 31;
  const bf16_t* row = baseT + (size_t)(n0 + (lane & 15)) * ld + k0
                    + ((lane < 16) ? 0 : 8);
  union { uint4 u[2]; v16bf v; } t;
  t.u[0] = *(const uint4*)(row);
  t.u[1] = *(const uint4*)(row + 16);
  return t.v;
}

// 16-lane-group cross-lane reductions (wave32; xor masks < 16 stay within the
// N-group of a C fragment).
__device__ __forceinline__ float rmax16(float v) {
#pragma unroll
  for (int m = 1; m < 16; m <<= 1) v = fmaxf(v, __shfl_xor(v, m, 32));
  return v;
}
__device__ __forceinline__ float rsum16(float v) {
#pragma unroll
  for (int m = 1; m < 16; m <<= 1) v += __shfl_xor(v, m, 32);
  return v;
}

// ---- conversion kernels ----------------------------------------------------
__global__ void cvt_bf16(const float* __restrict__ in, bf16_t* __restrict__ out,
                         int n) {
  int i = blockIdx.x * blockDim.x + threadIdx.x;
  if (i < n) out[i] = f2bf(in[i]);
}

// WT[n*D + k] = bf16(W[k*D + n])
__global__ void transpose_bf16(const float* __restrict__ W,
                               bf16_t* __restrict__ WT) {
  int i = blockIdx.x * blockDim.x + threadIdx.x;
  if (i < DD * DD) {
    int r = i >> 10;          // k
    int c = i & (DD - 1);     // n
    WT[(size_t)c * DD + r] = f2bf(W[i]);
  }
}

// ---- WMMA GEMM: C[M=8192, N=1024] = A[8192,1024] * B[1024,1024] ------------
// A bf16 row-major, BT bf16 N-major (transposed).  8 waves/block; each wave
// owns a 32x64 tile (2 A-frags x 4 B-frags -> 8 WMMAs per K step, B frags
// amortized over both M subtiles).  K-loop unrolled 4x: up to 48 B128 loads
// in flight (< 63 LOADcnt limit) to hide L2 latency behind the WMMA pipe.
// MODE 0: bf16 out [B,H,S,HD] (scaled by oscale); MODE 1: bf16 out [B,H,HD,S]
// (V transposed); MODE 2: f32 out [B,S,D] with bias.
template <int MODE>
__global__ void __launch_bounds__(256)
gemm_wmma(const bf16_t* __restrict__ A, const bf16_t* __restrict__ BT,
          void* __restrict__ out, const float* __restrict__ bias,
          float oscale) {
  int wave = threadIdx.x >> 5;
  int lane = threadIdx.x & 31;
  int m0 = (blockIdx.y * 8 + wave) * 32;
  int n0 = blockIdx.x * 64;

  v8f acc[2][4] = {};
#pragma unroll 4
  for (int k0 = 0; k0 < DD; k0 += 32) {
    if ((k0 & 127) == 0 && k0 + 128 < DD)
      __builtin_prefetch(A + (size_t)(m0 + (lane & 15)) * DD + k0 + 128, 0, 1);
    v16bf a0 = load_a_frag(A, DD, m0, k0);
    v16bf a1 = load_a_frag(A, DD, m0 + 16, k0);
#pragma unroll
    for (int j = 0; j < 4; ++j) {
      v16bf b = load_b_frag_t(BT, DD, k0, n0 + j * 16);
      acc[0][j] = __builtin_amdgcn_wmma_f32_16x16x32_bf16(
          false, a0, false, b, (short)0, acc[0][j], false, false);
      acc[1][j] = __builtin_amdgcn_wmma_f32_16x16x32_bf16(
          false, a1, false, b, (short)0, acc[1][j], false, false);
    }
  }

  int rbase = (lane < 16) ? 0 : 8;
  int nc = lane & 15;
#pragma unroll
  for (int t = 0; t < 2; ++t) {
#pragma unroll
    for (int j = 0; j < 4; ++j) {
#pragma unroll
      for (int r = 0; r < 8; ++r) {
        int m = m0 + t * 16 + rbase + r;
        int n = n0 + j * 16 + nc;
        float v = acc[t][j][r];
        if (MODE == 2) {
          ((float*)out)[(size_t)m * DD + n] = v + bias[n];
        } else {
          int b = m >> 11, s = m & (SS - 1);
          int h = n >> 6,  hd = n & (HDI - 1);
          if (MODE == 0)
            ((bf16_t*)out)[(((size_t)b * HH + h) * SS + s) * HDI + hd] =
                f2bf(v * oscale);
          else
            ((bf16_t*)out)[(((size_t)b * HH + h) * HDI + hd) * SS + s] =
                f2bf(v * oscale);
        }
      }
    }
  }
}

// ---- flash attention: one wave per (b, h, 16-query tile) -------------------
// Q (pre-scaled by 1/sqrt(HD)), K: bf16 [B,H,S,HD]; Vt: bf16 [B,H,HD,S];
// ctx out: bf16 [B,S,D].
__global__ void __launch_bounds__(256)
attn_fa(const bf16_t* __restrict__ Q, const bf16_t* __restrict__ K,
        const bf16_t* __restrict__ Vt, bf16_t* __restrict__ ctx) {
  __shared__ __align__(16) bf16_t pls[8][16 * 32];

  int wave = threadIdx.x >> 5;
  int lane = threadIdx.x & 31;
  int g = blockIdx.x * 8 + wave;          // 0 .. B*H*(S/16)-1
  int qt = g & (SS / 16 - 1);
  int bh = g >> 7;                        // S/16 == 128
  int qbase = qt * 16;

  const bf16_t* Qh = Q  + (size_t)bh * SS * HDI;
  const bf16_t* Kh = K  + (size_t)bh * SS * HDI;
  const bf16_t* Vh = Vt + (size_t)bh * HDI * SS;

  // Q fragments for the whole q-tile (HD=64 -> two k-steps of 32)
  v16bf aq0 = load_a_frag(Qh, HDI, qbase, 0);
  v16bf aq1 = load_a_frag(Qh, HDI, qbase, 32);

  v8f acc[4] = {};
  float mrow[8], lrow[8];
#pragma unroll
  for (int r = 0; r < 8; ++r) { mrow[r] = -1e30f; lrow[r] = 0.0f; }

  int rbase = (lane < 16) ? 0 : 8;
  int nc = lane & 15;
  bf16_t* myp = &pls[wave][0];

  for (int n0 = 0; n0 <= qbase + 15; n0 += 32) {
    v8f z = {};
    // scores for keys [n0, n0+16): contract over hd in two steps
    v16bf b00 = load_b_frag_t(Kh, HDI, 0,  n0);
    v16bf b01 = load_b_frag_t(Kh, HDI, 32, n0);
    v8f s0 = __builtin_amdgcn_wmma_f32_16x16x32_bf16(false, aq0, false, b00,
                                                     (short)0, z, false, false);
    s0 = __builtin_amdgcn_wmma_f32_16x16x32_bf16(false, aq1, false, b01,
                                                 (short)0, s0, false, false);
    // scores for keys [n0+16, n0+32)
    v16bf b10 = load_b_frag_t(Kh, HDI, 0,  n0 + 16);
    v16bf b11 = load_b_frag_t(Kh, HDI, 32, n0 + 16);
    v8f s1 = __builtin_amdgcn_wmma_f32_16x16x32_bf16(false, aq0, false, b10,
                                                     (short)0, z, false, false);
    s1 = __builtin_amdgcn_wmma_f32_16x16x32_bf16(false, aq1, false, b11,
                                                 (short)0, s1, false, false);

    // online softmax, row-by-row (each lane holds 8 rows x 1 column);
    // scores already scaled (Q pre-scaled by 1/sqrt(HD))
#pragma unroll
    for (int r = 0; r < 8; ++r) {
      int row = qbase + rbase + r;
      float s0r = (n0 + nc      <= row) ? s0[r] : -1e30f;
      float s1r = (n0 + 16 + nc <= row) ? s1[r] : -1e30f;
      float bm = rmax16(fmaxf(s0r, s1r));
      float mn = fmaxf(mrow[r], bm);
      float alpha = __expf(mrow[r] - mn);
      float p0 = __expf(s0r - mn);
      float p1 = __expf(s1r - mn);
      lrow[r] = lrow[r] * alpha + rsum16(p0 + p1);
      mrow[r] = mn;
      acc[0][r] *= alpha; acc[1][r] *= alpha;
      acc[2][r] *= alpha; acc[3][r] *= alpha;
      // stage P (C-frag layout) into LDS for A-frag reload
      myp[(rbase + r) * 32 + nc]      = f2bf(p0);
      myp[(rbase + r) * 32 + nc + 16] = f2bf(p1);
    }
    __builtin_amdgcn_wave_barrier();
    v16bf pa = load_a_frag(myp, 32, 0, 0);   // ds_load path (16x32 P tile)
#pragma unroll
    for (int j = 0; j < 4; ++j) {            // ctx += P * V, 64 hd columns
      v16bf bv = load_b_frag_t(Vh, SS, n0, j * 16);
      acc[j] = __builtin_amdgcn_wmma_f32_16x16x32_bf16(
          false, pa, false, bv, (short)0, acc[j], false, false);
    }
    __builtin_amdgcn_wave_barrier();
  }

  int b = bh >> 4, h = bh & (HH - 1);
#pragma unroll
  for (int r = 0; r < 8; ++r) {
    float inv = 1.0f / lrow[r];
    int row = qbase + rbase + r;
    size_t base = ((size_t)b * SS + row) * DD + h * HDI;
#pragma unroll
    for (int j = 0; j < 4; ++j)
      ctx[base + j * 16 + nc] = f2bf(acc[j][r] * inv);
  }
}

// ---------------------------------------------------------------------------
extern "C" void kernel_launch(void* const* d_in, const int* in_sizes, int n_in,
                              void* d_out, int out_size, void* d_ws,
                              size_t ws_size, hipStream_t stream) {
  (void)in_sizes; (void)n_in; (void)out_size; (void)ws_size;
  const float* x  = (const float*)d_in[0];
  const float* Wq = (const float*)d_in[1];
  const float* Wk = (const float*)d_in[2];
  const float* Wv = (const float*)d_in[3];
  const float* Wo = (const float*)d_in[4];
  const float* bo = (const float*)d_in[5];
  float* out = (float*)d_out;

  // workspace layout (bf16): xb(16M) WqT/WkT/WvT/WoT(2M ea) Qb/Kb/Vt/ctx(16M ea)
  char* ws = (char*)d_ws;
  size_t off = 0;
  auto alloc = [&](size_t bytes) -> char* {
    char* p = ws + off;
    off += (bytes + 255) & ~(size_t)255;
    return p;
  };
  const size_t NX = (size_t)BB * SS * DD;   // 8,388,608
  const size_t NW = (size_t)DD * DD;        // 1,048,576
  bf16_t* xb  = (bf16_t*)alloc(NX * 2);
  bf16_t* WqT = (bf16_t*)alloc(NW * 2);
  bf16_t* WkT = (bf16_t*)alloc(NW * 2);
  bf16_t* WvT = (bf16_t*)alloc(NW * 2);
  bf16_t* WoT = (bf16_t*)alloc(NW * 2);
  bf16_t* Qb  = (bf16_t*)alloc(NX * 2);
  bf16_t* Kb  = (bf16_t*)alloc(NX * 2);
  bf16_t* Vt  = (bf16_t*)alloc(NX * 2);
  bf16_t* cx  = (bf16_t*)alloc(NX * 2);

  cvt_bf16<<<(int)((NX + 255) / 256), 256, 0, stream>>>(x, xb, (int)NX);
  int tgrid = (int)((NW + 255) / 256);
  transpose_bf16<<<tgrid, 256, 0, stream>>>(Wq, WqT);
  transpose_bf16<<<tgrid, 256, 0, stream>>>(Wk, WkT);
  transpose_bf16<<<tgrid, 256, 0, stream>>>(Wv, WvT);
  transpose_bf16<<<tgrid, 256, 0, stream>>>(Wo, WoT);

  dim3 gg(DD / 64, (BB * SS) / 256);        // (16, 32): 32 rows x 8 waves
  gemm_wmma<0><<<gg, 256, 0, stream>>>(xb, WqT, Qb, nullptr, SCALE);
  gemm_wmma<0><<<gg, 256, 0, stream>>>(xb, WkT, Kb, nullptr, 1.0f);
  gemm_wmma<1><<<gg, 256, 0, stream>>>(xb, WvT, Vt, nullptr, 1.0f);

  attn_fa<<<(BB * HH * (SS / 16)) / 8, 256, 0, stream>>>(Qb, Kb, Vt, cx);

  gemm_wmma<2><<<gg, 256, 0, stream>>>(cx, WoT, out, bo, 1.0f);
}
